// LlamaAttention_57939108823331
// MI455X (gfx1250) — matrix operand order repaired and checked by reading, not compiled, not run on hardware
//
#include <hip/hip_runtime.h>
#include <hip/hip_bf16.h>

typedef __attribute__((ext_vector_type(16))) __bf16 v16bf;
typedef __attribute__((ext_vector_type(8)))  __bf16 v8bf;
typedef __attribute__((ext_vector_type(8)))  float  v8f;
typedef __attribute__((ext_vector_type(4)))  float  v4f;

#define DIM      4096
#define N_HEAD   32
#define N_KV     8
#define HD       128
#define SEQ      2048
#define QKV_DIM  6144   // (32 + 8 + 8) * 128

// ---------------------------------------------------------------------------
// helpers
// ---------------------------------------------------------------------------
__device__ __forceinline__ v8f vzero8f() {
    v8f z;
#pragma unroll
    for (int i = 0; i < 8; ++i) z[i] = 0.0f;
    return z;
}

// A-matrix 16x32 bf16 fragment from row-major LDS tile (ld halves per row).
// lane<16 : M=lane,    K = {0..7, 16..23}
// lane>=16: M=lane-16, K = {8..15, 24..31}
__device__ __forceinline__ v16bf frag_a(const __bf16* base, int ld, int lane) {
    int h = (lane >> 4) & 1;
    const __bf16* p = base + (lane & 15) * ld + h * 8;
    v8bf lo = *(const v8bf*)p;
    v8bf hi = *(const v8bf*)(p + 16);
    v16bf r;
#pragma unroll
    for (int i = 0; i < 8; ++i) { r[i] = lo[i]; r[i + 8] = hi[i]; }
    return r;
}

// B-matrix 32x16 bf16 fragment, built from row-major W[N][K] (so B = W^T).
// lane<16 : N=lane,    K = {0..7, 16..23}
// lane>=16: N=lane-16, K = {8..15, 24..31}
__device__ __forceinline__ v16bf frag_b(const __bf16* base, int ld, int lane) {
    int h = (lane >> 4) & 1;
    const __bf16* p = base + (lane & 15) * ld + h * 8;
    v8bf lo = *(const v8bf*)p;
    v8bf hi = *(const v8bf*)(p + 16);
    v16bf r;
#pragma unroll
    for (int i = 0; i < 8; ++i) { r[i] = lo[i]; r[i + 8] = hi[i]; }
    return r;
}

// ---------------------------------------------------------------------------
// fp32 -> bf16 convert (vectorized: 8 elems / thread / step)
// ---------------------------------------------------------------------------
__global__ void f32_to_bf16_k(const float* __restrict__ in,
                              __bf16* __restrict__ out, long n8) {
    long i = (long)blockIdx.x * blockDim.x + threadIdx.x;
    long stride = (long)gridDim.x * blockDim.x;
    for (; i < n8; i += stride) {
        v4f a = *(const v4f*)(in + i * 8);
        v4f b = *(const v4f*)(in + i * 8 + 4);
        v8bf o;
#pragma unroll
        for (int e = 0; e < 4; ++e) { o[e] = (__bf16)a[e]; o[e + 4] = (__bf16)b[e]; }
        *(v8bf*)(out + i * 8) = o;
    }
}

// ---------------------------------------------------------------------------
// GEMM: C[M,N] = A[M,K] * B[N,K]^T   (A,B bf16 row-major, C fp32)
// 128x256 block tile, 8 waves (256 thr), wave tile 64x64, BK=32,
// double-buffered LDS, global_prefetch of tile-after-next.
// ---------------------------------------------------------------------------
#define BM 128
#define BN 256
#define BK 32
#define LDT 40   // padded LDS row stride in halves

__global__ __launch_bounds__(256)
void gemm_bf16_nt(const __bf16* __restrict__ A, const __bf16* __restrict__ B,
                  float* __restrict__ C, int M, int N, int K) {
    __shared__ __attribute__((aligned(16))) __bf16 As[2][BM * LDT];
    __shared__ __attribute__((aligned(16))) __bf16 Bs[2][BN * LDT];

    int tid  = threadIdx.x;
    int lane = tid & 31;
    int wave = tid >> 5;
    int wm0  = (wave >> 2) * 64;   // 2 row-groups of 64
    int wn0  = (wave & 3) * 64;    // 4 col-groups of 64

    long m_blk = (long)blockIdx.y * BM;
    long n_blk = (long)blockIdx.x * BN;

    // staging assignments: A: 2 thr/row x 16 halves; B: 1 thr/row x 32 halves
    int arow = tid >> 1, acol = (tid & 1) * 16;
    int brow = tid;
    const __bf16* Ag = A + (m_blk + arow) * (long)K + acol;
    const __bf16* Bg = B + (n_blk + brow) * (long)K;

    v8f acc[4][4];
#pragma unroll
    for (int i = 0; i < 4; ++i)
#pragma unroll
        for (int j = 0; j < 4; ++j) acc[i][j] = vzero8f();

    // ---- preload k-tile 0 into buffer 0 ----
    {
        *(v8bf*)&As[0][arow * LDT + acol]     = *(const v8bf*)(Ag);
        *(v8bf*)&As[0][arow * LDT + acol + 8] = *(const v8bf*)(Ag + 8);
#pragma unroll
        for (int c = 0; c < 4; ++c)
            *(v8bf*)&Bs[0][brow * LDT + c * 8] = *(const v8bf*)(Bg + c * 8);
    }
    __syncthreads();

    int KT = K / BK;
    for (int kt = 0; kt < KT; ++kt) {
        int cur = kt & 1, nxt = cur ^ 1;
        bool more = (kt + 1) < KT;

        // issue global loads for next tile early (hidden under WMMA below)
        v8bf na0, na1, nb[4];
        if (more) {
            long kb = (long)(kt + 1) * BK;
            na0 = *(const v8bf*)(Ag + kb);
            na1 = *(const v8bf*)(Ag + kb + 8);
#pragma unroll
            for (int c = 0; c < 4; ++c)
                nb[c] = *(const v8bf*)(Bg + kb + c * 8);
            if (kt + 2 < KT) {   // warm GL2 for the tile after next
                __builtin_prefetch(Ag + kb + BK, 0, 1);
                __builtin_prefetch(Bg + kb + BK, 0, 1);
            }
        }

        // ---- compute from current buffer ----
        v16bf af[4], bfr[4];
#pragma unroll
        for (int i = 0; i < 4; ++i)
            af[i] = frag_a(&As[cur][(wm0 + i * 16) * LDT], LDT, lane);
#pragma unroll
        for (int j = 0; j < 4; ++j)
            bfr[j] = frag_b(&Bs[cur][(wn0 + j * 16) * LDT], LDT, lane);
#pragma unroll
        for (int i = 0; i < 4; ++i)
#pragma unroll
            for (int j = 0; j < 4; ++j)
                acc[i][j] = __builtin_amdgcn_wmma_f32_16x16x32_bf16(
                    false, af[i], false, bfr[j], (short)0, acc[i][j], false, false);

        // ---- stage next tile into the other buffer ----
        if (more) {
            *(v8bf*)&As[nxt][arow * LDT + acol]     = na0;
            *(v8bf*)&As[nxt][arow * LDT + acol + 8] = na1;
#pragma unroll
            for (int c = 0; c < 4; ++c)
                *(v8bf*)&Bs[nxt][brow * LDT + c * 8] = nb[c];
        }
        __syncthreads();
    }

    int hf = lane >> 4, nn = lane & 15;
#pragma unroll
    for (int i = 0; i < 4; ++i)
#pragma unroll
        for (int j = 0; j < 4; ++j)
#pragma unroll
            for (int r = 0; r < 8; ++r) {
                long mm = m_blk + wm0 + i * 16 + r + 8 * hf;
                long cc = n_blk + wn0 + j * 16 + nn;
                C[mm * (long)N + cc] = acc[i][j][r];
            }
}

// ---------------------------------------------------------------------------
// RoPE + split qkv -> head-major bf16 Q/K/V
// grid: (48, SEQ) blocks of 64 threads
// ---------------------------------------------------------------------------
__global__ __launch_bounds__(64)
void rope_split(const float* __restrict__ qkv, __bf16* __restrict__ qr,
                __bf16* __restrict__ kr, __bf16* __restrict__ vr) {
    int h = blockIdx.x;    // 0..47: 32 q heads, 8 k heads, 8 v heads
    int s = blockIdx.y;    // 0..SEQ-1
    int d = threadIdx.x;   // 0..63
    const float* rowp = qkv + (long)s * QKV_DIM;

    float ang = (float)s * __powf(10000.0f, -(float)d * (1.0f / 64.0f));
    float c  = __cosf(ang);
    float sn = __sinf(ang);

    if (h < N_HEAD) {
        const float* p = rowp + h * HD;
        float a = p[d], b = p[d + 64];
        __bf16* o = qr + ((long)h * SEQ + s) * HD;
        o[d]      = (__bf16)(a * c - b * sn);
        o[d + 64] = (__bf16)(b * c + a * sn);
    } else if (h < N_HEAD + N_KV) {
        int kvh = h - N_HEAD;
        const float* p = rowp + DIM + kvh * HD;
        float a = p[d], b = p[d + 64];
        __bf16* o = kr + ((long)kvh * SEQ + s) * HD;
        o[d]      = (__bf16)(a * c - b * sn);
        o[d + 64] = (__bf16)(b * c + a * sn);
    } else {
        int kvh = h - N_HEAD - N_KV;
        const float* p = rowp + DIM + N_KV * HD + kvh * HD;
        __bf16* o = vr + ((long)kvh * SEQ + s) * HD;
        o[d]      = (__bf16)p[d];
        o[d + 64] = (__bf16)p[d + 64];
    }
}

// ---------------------------------------------------------------------------
// Flash attention: per (q-block of 64 rows, head); 4 waves / 128 threads
// QK^T and P*V via WMMA bf16; online softmax; causal mask analytic.
// grid: (SEQ/64, N_HEAD)
// ---------------------------------------------------------------------------
#define LQ 136   // 128 + 8 pad, halves
#define LV 72    // 64 + 8 pad, halves

__global__ __launch_bounds__(128)
void attn_fa(const __bf16* __restrict__ Q, const __bf16* __restrict__ Kr,
             const __bf16* __restrict__ V, __bf16* __restrict__ Y) {
    __shared__ __attribute__((aligned(16))) __bf16 Qs[64 * LQ];
    __shared__ __attribute__((aligned(16))) __bf16 Ks[64 * LQ];
    __shared__ __attribute__((aligned(16))) __bf16 Vt[128 * LV]; // V transposed: Vt[d][k]
    __shared__ __attribute__((aligned(16))) __bf16 Ps[64 * LV];  // probs, bf16

    int tid = threadIdx.x, lane = tid & 31, wave = tid >> 5;
    int head = blockIdx.y, qb = blockIdx.x;
    int kvh = head >> 2;          // GQA: 4 q heads per kv head
    int q0  = qb * 64;
    int m0  = wave * 16;          // this wave's 16-row strip
    const float scale = 0.08838834764831845f; // 1/sqrt(128)

    // load Q block (64 x 128)
    for (int idx = tid; idx < 64 * 16; idx += 128) {
        int r = idx >> 4, c = (idx & 15) * 8;
        *(v8bf*)&Qs[r * LQ + c] =
            *(const v8bf*)&Q[((long)head * SEQ + q0 + r) * HD + c];
    }

    float mprev[8], lsum[8];
    v8f oacc[8];
#pragma unroll
    for (int r = 0; r < 8; ++r) { mprev[r] = -3.0e38f; lsum[r] = 0.0f; }
#pragma unroll
    for (int t = 0; t < 8; ++t) oacc[t] = vzero8f();

    int hf = lane >> 4, nn = lane & 15;

    for (int kb = 0; kb <= qb; ++kb) {
        __syncthreads();   // protect previous iteration's Ks/Vt readers
        int kbase = kb * 64;
        for (int idx = tid; idx < 64 * 16; idx += 128) {
            int r = idx >> 4, c = (idx & 15) * 8;
            long src = ((long)kvh * SEQ + kbase + r) * HD + c;
            *(v8bf*)&Ks[r * LQ + c] = *(const v8bf*)&Kr[src];
            v8bf vv = *(const v8bf*)&V[src];
#pragma unroll
            for (int e = 0; e < 8; ++e) Vt[(c + e) * LV + r] = vv[e];
        }
        if (kb < qb) {     // warm GL2 for next K/V block
            long nsrc = ((long)kvh * SEQ + kbase + 64 + (tid & 63)) * HD + (tid >> 6) * 64;
            __builtin_prefetch(&Kr[nsrc], 0, 1);
            __builtin_prefetch(&V[nsrc], 0, 1);
        }
        __syncthreads();

        // ---- S = Q * K^T (16 x 64 strip per wave), K-dim = 128 ----
        v8f sacc[4];
#pragma unroll
        for (int t = 0; t < 4; ++t) sacc[t] = vzero8f();
#pragma unroll
        for (int kk = 0; kk < 4; ++kk) {
            v16bf aq = frag_a(&Qs[m0 * LQ + kk * 32], LQ, lane);
#pragma unroll
            for (int t = 0; t < 4; ++t) {
                v16bf bk = frag_b(&Ks[(t * 16) * LQ + kk * 32], LQ, lane);
                sacc[t] = __builtin_amdgcn_wmma_f32_16x16x32_bf16(
                    false, aq, false, bk, (short)0, sacc[t], false, false);
            }
        }

        // ---- online softmax (per row: reduce over 4 tiles x 16 lanes) ----
#pragma unroll
        for (int r = 0; r < 8; ++r) {
            int mrow = m0 + r + 8 * hf;
            int qabs = q0 + mrow;
            float sv[4];
            float vmax = -3.0e38f;
#pragma unroll
            for (int t = 0; t < 4; ++t) {
                int kabs = kbase + t * 16 + nn;
                float s = sacc[t][r] * scale;
                if (kabs > qabs) s = -1.0e9f;   // causal mask
                sv[t] = s;
                vmax = fmaxf(vmax, s);
            }
#pragma unroll
            for (int msk = 1; msk < 16; msk <<= 1)
                vmax = fmaxf(vmax, __shfl_xor(vmax, msk, 32));
            float mnew = fmaxf(mprev[r], vmax);
            float f = __expf(mprev[r] - mnew);
            float rs = 0.0f;
#pragma unroll
            for (int t = 0; t < 4; ++t) {
                float p = __expf(sv[t] - mnew);
                Ps[mrow * LV + t * 16 + nn] = (__bf16)p;
                rs += p;
            }
#pragma unroll
            for (int msk = 1; msk < 16; msk <<= 1)
                rs += __shfl_xor(rs, msk, 32);
            lsum[r] = lsum[r] * f + rs;
            mprev[r] = mnew;
#pragma unroll
            for (int td = 0; td < 8; ++td) oacc[td][r] *= f;
        }

        // ---- O += P * V  (16 x 128 strip per wave), K-dim = 64 ----
        v16bf ap0 = frag_a(&Ps[m0 * LV + 0],  LV, lane);
        v16bf ap1 = frag_a(&Ps[m0 * LV + 32], LV, lane);
#pragma unroll
        for (int td = 0; td < 8; ++td) {
            v16bf bv0 = frag_b(&Vt[(td * 16) * LV + 0],  LV, lane);
            oacc[td] = __builtin_amdgcn_wmma_f32_16x16x32_bf16(
                false, ap0, false, bv0, (short)0, oacc[td], false, false);
            v16bf bv1 = frag_b(&Vt[(td * 16) * LV + 32], LV, lane);
            oacc[td] = __builtin_amdgcn_wmma_f32_16x16x32_bf16(
                false, ap1, false, bv1, (short)0, oacc[td], false, false);
        }
    }

    // ---- epilogue: normalize and store y (bf16, [s][h*128+d]) ----
#pragma unroll
    for (int td = 0; td < 8; ++td)
#pragma unroll
        for (int r = 0; r < 8; ++r) {
            int mrow = m0 + r + 8 * hf;
            float o = oacc[td][r] / lsum[r];
            Y[((long)(q0 + mrow)) * DIM + head * HD + td * 16 + nn] = (__bf16)o;
        }
}

// ---------------------------------------------------------------------------
// launch
// ---------------------------------------------------------------------------
extern "C" void kernel_launch(void* const* d_in, const int* in_sizes, int n_in,
                              void* d_out, int out_size, void* d_ws, size_t ws_size,
                              hipStream_t stream) {
    const float* x    = (const float*)d_in[0];   // [1, 2048, 4096]
    const float* wqkv = (const float*)d_in[1];   // [6144, 4096]
    const float* wo   = (const float*)d_in[2];   // [4096, 4096]
    // d_in[3] = mask: causal, reproduced analytically in attn_fa

    char* ws = (char*)d_ws;
    size_t off = 0;
    auto take = [&](size_t bytes) -> void* {
        void* p = (void*)(ws + off);
        off += (bytes + 255) & ~(size_t)255;
        return p;
    };

    __bf16* xb    = (__bf16*)take((size_t)SEQ * DIM * 2);          // 16 MB
    __bf16* wqkvb = (__bf16*)take((size_t)QKV_DIM * DIM * 2);      // 48 MB
    __bf16* wob   = (__bf16*)take((size_t)DIM * DIM * 2);          // 32 MB
    float*  qkv   = (float*) take((size_t)SEQ * QKV_DIM * 4);      // 48 MB
    __bf16* qr    = (__bf16*)take((size_t)N_HEAD * SEQ * HD * 2);  // 16 MB
    __bf16* kr    = (__bf16*)take((size_t)N_KV * SEQ * HD * 2);    //  4 MB
    __bf16* vr    = (__bf16*)take((size_t)N_KV * SEQ * HD * 2);    //  4 MB
    __bf16* yb    = (__bf16*)take((size_t)SEQ * DIM * 2);          // 16 MB

    // 1) convert fp32 inputs to bf16 (8 elems / thread / step)
    f32_to_bf16_k<<<1024, 256, 0, stream>>>(x, xb, (long)SEQ * DIM / 8);
    f32_to_bf16_k<<<1024, 256, 0, stream>>>(wqkv, wqkvb, (long)QKV_DIM * DIM / 8);
    f32_to_bf16_k<<<1024, 256, 0, stream>>>(wo, wob, (long)DIM * DIM / 8);

    // 2) qkv = x * wqkv^T    [2048, 6144]
    gemm_bf16_nt<<<dim3(QKV_DIM / BN, SEQ / BM), 256, 0, stream>>>(
        xb, wqkvb, qkv, SEQ, QKV_DIM, DIM);

    // 3) RoPE + split into head-major bf16 Q/K/V
    rope_split<<<dim3(N_HEAD + 2 * N_KV, SEQ), 64, 0, stream>>>(qkv, qr, kr, vr);

    // 4) flash attention -> y bf16 [2048, 4096]
    attn_fa<<<dim3(SEQ / 64, N_HEAD), 128, 0, stream>>>(qr, kr, vr, yb);

    // 5) out = y * wo^T      [2048, 4096] fp32
    gemm_bf16_nt<<<dim3(DIM / BN, SEQ / BM), 256, 0, stream>>>(
        yb, wob, (float*)d_out, SEQ, DIM, DIM);
}
